// LoadingListEncoder_62466004353385
// MI455X (gfx1250) — matrix operand order repaired and checked by reading, not compiled, not run on hardware
//
#include <hip/hip_runtime.h>
#include <stdint.h>

#define VOCAB 1000
#define EMB   128
#define HID   256
#define GATES (4 * HID)   // 1024
#define BATCH 64
#define SEQT  1024

// dynamic LDS layout (bytes)
#define GATES_BYTES (BATCH * GATES * 4)         // 262144
#define X_OFF       GATES_BYTES                 // 262144
#define X_BYTES     (BATCH * EMB * 2)           // 16384
#define H_OFF       (X_OFF + X_BYTES)           // 278528
#define H_BYTES     (BATCH * HID * 2)           // 32768
#define BSUM_OFF    (H_OFF + H_BYTES)           // 311296
#define SMEM_TOTAL  (BSUM_OFF + GATES * 4)      // 315392  (<= 320KB WGP LDS)

typedef __bf16 bf16_t;
typedef bf16_t v16bf __attribute__((ext_vector_type(16)));
typedef float  v8f   __attribute__((ext_vector_type(8)));

union FragAB { uint4 q[2]; v16bf v; };

__device__ __forceinline__ unsigned short f2bf(float f) {
  union { float f; unsigned int u; } x;
  x.f = f;
  unsigned int u = x.u;
  unsigned int r = u + 0x7FFFu + ((u >> 16) & 1u);   // round-to-nearest-even
  return (unsigned short)(r >> 16);
}

// ---------------------------------------------------------------------------
// Prep: W_ih [4H][E] -> WihT bf16 [E][4H];  W_hh [4H][H] -> WhhT bf16 [H][4H]
// Transposed K-major layout: each lane's WMMA B-fragment is 16 bf16 loads
// strided by 4H (lane&15 = N column, 16 K rows per lane).
// ---------------------------------------------------------------------------
__global__ void prep_weights(const float* __restrict__ Wih,
                             const float* __restrict__ Whh,
                             unsigned short* __restrict__ WihT,
                             unsigned short* __restrict__ WhhT) {
  int idx = blockIdx.x * blockDim.x + threadIdx.x;
  if (idx < EMB * GATES) {
    int k = idx >> 10;         // 0..127
    int n = idx & 1023;
    WihT[idx] = f2bf(Wih[n * EMB + k]);
  } else {
    int i2 = idx - EMB * GATES;
    if (i2 < HID * GATES) {
      int k = i2 >> 10;        // 0..255
      int n = i2 & 1023;
      WhhT[i2] = f2bf(Whh[n * HID + k]);
    }
  }
}

// ---------------------------------------------------------------------------
// Persistent fused LSTM: one workgroup (32 wave32s on one WGP) owns the whole
// sequential recurrence. Per step: gather embeddings -> bf16 LDS; 256 gate
// tiles (16x16, K=128(x)+256(h)) via v_wmma_f32_16x16x32_bf16; barrier;
// masked element-wise update (c,h fp32 in regs; h re-packed bf16 into LDS).
// ---------------------------------------------------------------------------
__global__ __launch_bounds__(1024, 1) void lstm_persistent(
    const int* __restrict__ tokens,           // [B][T]
    const int* __restrict__ lengths,          // [B]
    const float* __restrict__ emb,            // [VOCAB][E]
    const unsigned short* __restrict__ WihT,  // [E][4H] bf16
    const unsigned short* __restrict__ WhhT,  // [H][4H] bf16
    const float* __restrict__ b_ih,           // [4H]
    const float* __restrict__ b_hh,           // [4H]
    float* __restrict__ out)                  // [B][H]
{
  extern __shared__ char smem[];
  float*          gates = (float*)smem;                         // [64][1024] f32
  unsigned short* xs    = (unsigned short*)(smem + X_OFF);      // [64][128] bf16
  unsigned short* hs    = (unsigned short*)(smem + H_OFF);      // [64][256] bf16
  float*          bsum  = (float*)(smem + BSUM_OFF);            // [1024] f32

  const int tid  = threadIdx.x;
  const int lane = tid & 31;
  const int wave = tid >> 5;

  // element-wise / loader mapping: thread -> (batch row b, 16-col segment)
  const int b   = tid >> 4;          // 0..63
  const int seg = tid & 15;          // 0..15
  const int j0  = seg * 16;

  // WMMA tile mapping: wave -> M block (16 rows of B) x 8 N-tiles
  const int m0   = (wave >> 3) << 4;        // 0,16,32,48
  const int ngrp = wave & 7;                // 8 groups of 8 N-tiles
  const int arow = m0 + (lane & 15);        // A-frag source row
  const int ahi  = (lane >> 4) * 8;         // K sub-offset per lane half

  // ---- init ----
  bsum[tid] = b_ih[tid] + b_hh[tid];        // fold both biases once
  #pragma unroll
  for (int k = 0; k < 16; ++k) hs[b * HID + j0 + k] = 0;
  float creg[16], hreg[16];
  #pragma unroll
  for (int k = 0; k < 16; ++k) { creg[k] = 0.f; hreg[k] = 0.f; }
  const int mylen = lengths[b];

  __syncthreads();

  for (int t = 0; t < SEQT; ++t) {
    // ---- phase L: x_t = bf16(emb[token]) into LDS ----
    {
      int token = tokens[b * SEQT + t];
      const float4* e = (const float4*)(emb + token * EMB + seg * 8);
      float4 f0 = e[0], f1 = e[1];
      union { unsigned short s[8]; uint4 q; } pk;
      pk.s[0] = f2bf(f0.x); pk.s[1] = f2bf(f0.y);
      pk.s[2] = f2bf(f0.z); pk.s[3] = f2bf(f0.w);
      pk.s[4] = f2bf(f1.x); pk.s[5] = f2bf(f1.y);
      pk.s[6] = f2bf(f1.z); pk.s[7] = f2bf(f1.w);
      *(uint4*)(xs + b * EMB + seg * 8) = pk.q;
    }
    __syncthreads();  // x ready; previous step's h writes visible

    // ---- phase G: gates = [x | h] @ [WihT ; WhhT]  (bf16 WMMA, f32 acc) ----
    for (int q = 0; q < 8; ++q) {
      const int n0 = ((ngrp << 3) + q) << 4;       // 0..1008
      // warm L2/L0 for this tile's WhhT rows
      __builtin_prefetch(WhhT + lane * GATES + n0, 0, 1);
      v8f acc = {0.f, 0.f, 0.f, 0.f, 0.f, 0.f, 0.f, 0.f};
      #pragma unroll
      for (int k0 = 0; k0 < EMB; k0 += 32) {       // input-projection part
        FragAB a, bb;
        a.q[0] = *(const uint4*)(xs + arow * EMB + k0 + ahi);
        a.q[1] = *(const uint4*)(xs + arow * EMB + k0 + ahi + 16);
        const unsigned short* bp = WihT + (k0 + ahi) * GATES + n0 + (lane & 15);
        for (int s = 0; s < 16; ++s) ((unsigned short*)bb.q)[s] = bp[(s + (s & 8)) * GATES];
        acc = __builtin_amdgcn_wmma_f32_16x16x32_bf16(
            false, a.v, false, bb.v, (short)0, acc, false, false);
      }
      #pragma unroll
      for (int k0 = 0; k0 < HID; k0 += 32) {       // recurrent part
        FragAB a, bb;
        a.q[0] = *(const uint4*)(hs + arow * HID + k0 + ahi);
        a.q[1] = *(const uint4*)(hs + arow * HID + k0 + ahi + 16);
        const unsigned short* bp = WhhT + (k0 + ahi) * GATES + n0 + (lane & 15);
        for (int s = 0; s < 16; ++s) ((unsigned short*)bb.q)[s] = bp[(s + (s & 8)) * GATES];
        acc = __builtin_amdgcn_wmma_f32_16x16x32_bf16(
            false, a.v, false, bb.v, (short)0, acc, false, false);
      }
      // D layout: VGPR r -> M = r (+8 for upper lane half), lane&15 -> N
      #pragma unroll
      for (int r = 0; r < 8; ++r) {
        gates[(m0 + r + ((lane >> 4) << 3)) * GATES + n0 + (lane & 15)] = acc[r];
      }
    }
    __syncthreads();  // gates ready

    // ---- phase U: masked LSTM cell update (i,f,g,o gate order) ----
    if (t < mylen) {
      const float* grow = gates + b * GATES;
      #pragma unroll
      for (int k = 0; k < 16; ++k) {
        int j = j0 + k;
        float ig = grow[j]           + bsum[j];
        float fg = grow[HID + j]     + bsum[HID + j];
        float gg = grow[2 * HID + j] + bsum[2 * HID + j];
        float og = grow[3 * HID + j] + bsum[3 * HID + j];
        ig = 1.f / (1.f + __expf(-ig));
        fg = 1.f / (1.f + __expf(-fg));
        og = 1.f / (1.f + __expf(-og));
        gg = tanhf(gg);
        float c = fg * creg[k] + ig * gg;
        creg[k] = c;
        float h = og * tanhf(c);
        hreg[k] = h;
        hs[b * HID + j] = f2bf(h);   // frozen (not rewritten) when t >= len
      }
    }
  }

  #pragma unroll
  for (int k = 0; k < 16; ++k) out[b * HID + j0 + k] = hreg[k];
}

// ---------------------------------------------------------------------------
extern "C" void kernel_launch(void* const* d_in, const int* in_sizes, int n_in,
                              void* d_out, int out_size, void* d_ws, size_t ws_size,
                              hipStream_t stream) {
  const int*   tokens  = (const int*)d_in[0];
  const int*   lengths = (const int*)d_in[1];
  const float* emb     = (const float*)d_in[2];
  const float* W_ih    = (const float*)d_in[3];
  const float* W_hh    = (const float*)d_in[4];
  const float* b_ih    = (const float*)d_in[5];
  const float* b_hh    = (const float*)d_in[6];
  float*       out     = (float*)d_out;

  unsigned short* WihT = (unsigned short*)d_ws;                       // 256 KB
  unsigned short* WhhT = (unsigned short*)((char*)d_ws + EMB * GATES * 2); // 512 KB

  // allow > default dynamic LDS (308 KB of the WGP's 320 KB)
  (void)hipFuncSetAttribute((const void*)lstm_persistent,
                            hipFuncAttributeMaxDynamicSharedMemorySize,
                            SMEM_TOTAL);

  int total = EMB * GATES + HID * GATES;   // 393216
  prep_weights<<<(total + 255) / 256, 256, 0, stream>>>(W_ih, W_hh, WihT, WhhT);

  lstm_persistent<<<1, 1024, SMEM_TOTAL, stream>>>(
      tokens, lengths, emb, WihT, WhhT, b_ih, b_hh, out);
}
